// FeedbackModelMPL_74311524155792
// MI455X (gfx1250) — matrix operand order repaired and checked by reading, not compiled, not run on hardware
//
#include <hip/hip_runtime.h>

// ---------------- problem constants ----------------
#define B_    16
#define S_    1024
#define H_    1024
#define NS_   32
#define HEADS_ 16
#define DH_   64
#define L_    5
#define G4H_  4096   // 4*H
#define LN_EPS_ 1e-7f

// ---------------- WMMA types ----------------
typedef __attribute__((ext_vector_type(16))) __bf16 v16bf;
typedef __attribute__((ext_vector_type(8)))  float  v8f;
typedef __attribute__((ext_vector_type(4)))  unsigned int u32x4;

union FragBF {
  v16bf v;
  u32x4 u[2];
};

__device__ __forceinline__ unsigned short f32_to_bf16_rne(float f) {
  unsigned int u = __float_as_uint(f);
  unsigned int r = 0x7FFFu + ((u >> 16) & 1u);
  return (unsigned short)((u + r) >> 16);
}

// A-operand fragment (M x K = 16 x 32), per ISA 16-bit A layout:
//  lanes 0-15 : row M=lane,  K = {0..7} U {16..23}
//  lanes16-31 : row M=lane-16, K = {8..15} U {24..31}
__device__ __forceinline__ v16bf load_frag_A(const unsigned short* base,
                                             int row0, int k0, int ld) {
  int lane = threadIdx.x & 31;
  int hi   = lane >> 4;            // 0 or 1
  int row  = row0 + (lane & 15);
  const unsigned short* p = base + (size_t)row * ld + k0 + hi * 8;
  FragBF f;
  f.u[0] = *(const u32x4*)(p);          // K chunk 0 (8 bf16)
  f.u[1] = *(const u32x4*)(p + 16);     // K chunk 1 (8 bf16)
  return f.v;
}

// B-operand fragment (K x N = 32 x 16) with B = W^T, W row-major [N,K]:
//  lane holds column n = lane&15, K = (lane>>4)*8 + {0..7} U {16..23}
__device__ __forceinline__ v16bf load_frag_B(const unsigned short* w,
                                             int n0, int k0, int ld) {
  int lane = threadIdx.x & 31;
  int n    = n0 + (lane & 15);
  const unsigned short* p = w + (size_t)n * ld + k0 + (lane >> 4) * 8;
  FragBF f;
  f.u[0] = *(const u32x4*)(p);
  f.u[1] = *(const u32x4*)(p + 16);
  return f.v;
}

// ---------------- conversion / init ----------------
__global__ void k_f32_to_bf16(const float* __restrict__ in,
                              unsigned short* __restrict__ out, int n) {
  int i = blockIdx.x * 256 + threadIdx.x;
  if (i < n) out[i] = f32_to_bf16_rne(in[i]);
}

__global__ void k_init_state(float* __restrict__ c, unsigned short* __restrict__ h0, int n) {
  int i = blockIdx.x * 256 + threadIdx.x;
  if (i < n) { c[i] = 0.f; h0[i] = 0; }
}

// ---------------- generic bf16 WMMA GEMM:  D[M,N] = A[M,K] @ W[N,K]^T (+bias) -----
// block = 128 threads (4 waves); wave computes a 16x64 tile; block covers 16x256.
// Explicit 2x-unrolled ping-pong double buffering: two disjoint fragment register
// sets alternate, so no register copies are needed and each set's reload hides
// behind the other set's 4-WMMA group.
// Requires K % 64 == 0 (true for all uses here: K = 1024).
__global__ __launch_bounds__(128)
void k_gemm_bf16_nt(const unsigned short* __restrict__ A,
                    const unsigned short* __restrict__ W,
                    float* __restrict__ D, const float* __restrict__ bias,
                    int M, int N, int K) {
  int wave = threadIdx.x >> 5;
  int lane = threadIdx.x & 31;
  int m0 = blockIdx.y * 16;
  int n0 = blockIdx.x * 256 + wave * 64;
  if (m0 >= M || n0 >= N) return;

  v8f acc[4] = {};
  v16bf a0 = load_frag_A(A, m0, 0, K);
  v16bf b0[4];
#pragma unroll
  for (int t = 0; t < 4; ++t) b0[t] = load_frag_B(W, n0 + t * 16, 0, K);
  v16bf a1 = load_frag_A(A, m0, 32, K);
  v16bf b1[4];
#pragma unroll
  for (int t = 0; t < 4; ++t) b1[t] = load_frag_B(W, n0 + t * 16, 32, K);

  int kiter = K >> 5;                   // even by contract
#pragma unroll 1
  for (int i = 0; i < kiter - 2; i += 2) {
    int k2 = (i + 2) << 5;
    int k3 = (i + 3) << 5;
    if (k3 + 32 < K)                    // L1 prefetch two sets ahead (speculative)
      __builtin_prefetch(W + (size_t)(n0 + lane) * K + k3 + 32, 0, 1);
#pragma unroll
    for (int t = 0; t < 4; ++t)
      acc[t] = __builtin_amdgcn_wmma_f32_16x16x32_bf16(
          false, a0, false, b0[t], (short)0, acc[t], false, false);
    a0 = load_frag_A(A, m0, k2, K);
#pragma unroll
    for (int t = 0; t < 4; ++t) b0[t] = load_frag_B(W, n0 + t * 16, k2, K);
#pragma unroll
    for (int t = 0; t < 4; ++t)
      acc[t] = __builtin_amdgcn_wmma_f32_16x16x32_bf16(
          false, a1, false, b1[t], (short)0, acc[t], false, false);
    a1 = load_frag_A(A, m0, k3, K);
#pragma unroll
    for (int t = 0; t < 4; ++t) b1[t] = load_frag_B(W, n0 + t * 16, k3, K);
  }
#pragma unroll
  for (int t = 0; t < 4; ++t)
    acc[t] = __builtin_amdgcn_wmma_f32_16x16x32_bf16(
        false, a0, false, b0[t], (short)0, acc[t], false, false);
#pragma unroll
  for (int t = 0; t < 4; ++t)
    acc[t] = __builtin_amdgcn_wmma_f32_16x16x32_bf16(
        false, a1, false, b1[t], (short)0, acc[t], false, false);

  int half = lane >> 4, nin = lane & 15;
#pragma unroll
  for (int t = 0; t < 4; ++t) {
    int n = n0 + t * 16 + nin;
    float bvv = bias ? bias[n] : 0.f;
#pragma unroll
    for (int r = 0; r < 8; ++r) {
      int m = m0 + r + half * 8;
      D[(size_t)m * N + n] = acc[t][r] + bvv;
    }
  }
}

// ---------------- fused LSTM timestep -------------------------------------------
// grid = 8 blocks; block jb owns hidden slice j in [jb*128, jb*128+128).
// 256 threads = 8 waves; wave w: gate = w&3, jhalf = w>>2; computes gates via WMMA
// (A = h_bf ping buffer, 16x1024 bf16; W = W_hh bf16), stages [16 x 512] f32 in LDS,
// then pointwise i/f/g/o update; writes enc (f32) and next h (bf16 pong buffer).
// Same explicit ping-pong double buffering; this GEMM sits on the 1024-step serial
// recurrence chain so hiding L2 latency here is the whole kernel's critical path.
__global__ __launch_bounds__(256)
void k_lstm_step(const unsigned short* __restrict__ h_in,
                 const unsigned short* __restrict__ Whh,
                 const float* __restrict__ Xg,
                 const float* __restrict__ b_ih, const float* __restrict__ b_hh,
                 float* __restrict__ c_st, float* __restrict__ enc,
                 unsigned short* __restrict__ h_out, int t) {
  __shared__ float lds[16 * 512];   // rows = batch, cols = gate*128 + jlocal
  int jb   = blockIdx.x;            // 0..7
  int wave = threadIdx.x >> 5;      // 0..7
  int lane = threadIdx.x & 31;
  int gate = wave & 3;
  int jh   = wave >> 2;             // 0..1
  int lbase = gate * 128 + jh * 64;
  int nbase = gate * 1024 + jb * 128 + jh * 64;   // column into [4096]

  v8f acc[4] = {};
  v16bf a0 = load_frag_A(h_in, 0, 0, H_);
  v16bf b0[4];
#pragma unroll
  for (int s = 0; s < 4; ++s) b0[s] = load_frag_B(Whh, nbase + s * 16, 0, H_);
  v16bf a1 = load_frag_A(h_in, 0, 32, H_);
  v16bf b1[4];
#pragma unroll
  for (int s = 0; s < 4; ++s) b1[s] = load_frag_B(Whh, nbase + s * 16, 32, H_);

#pragma unroll 1
  for (int i = 0; i < (H_ >> 5) - 2; i += 2) {
    int k2 = (i + 2) << 5;
    int k3 = (i + 3) << 5;
    if (k3 + 32 < H_)
      __builtin_prefetch(Whh + (size_t)(nbase + lane) * H_ + k3 + 32, 0, 1);
#pragma unroll
    for (int s = 0; s < 4; ++s)
      acc[s] = __builtin_amdgcn_wmma_f32_16x16x32_bf16(
          false, a0, false, b0[s], (short)0, acc[s], false, false);
    a0 = load_frag_A(h_in, 0, k2, H_);
#pragma unroll
    for (int s = 0; s < 4; ++s) b0[s] = load_frag_B(Whh, nbase + s * 16, k2, H_);
#pragma unroll
    for (int s = 0; s < 4; ++s)
      acc[s] = __builtin_amdgcn_wmma_f32_16x16x32_bf16(
          false, a1, false, b1[s], (short)0, acc[s], false, false);
    a1 = load_frag_A(h_in, 0, k3, H_);
#pragma unroll
    for (int s = 0; s < 4; ++s) b1[s] = load_frag_B(Whh, nbase + s * 16, k3, H_);
  }
#pragma unroll
  for (int s = 0; s < 4; ++s)
    acc[s] = __builtin_amdgcn_wmma_f32_16x16x32_bf16(
        false, a0, false, b0[s], (short)0, acc[s], false, false);
#pragma unroll
  for (int s = 0; s < 4; ++s)
    acc[s] = __builtin_amdgcn_wmma_f32_16x16x32_bf16(
        false, a1, false, b1[s], (short)0, acc[s], false, false);

  int half = lane >> 4, nin = lane & 15;
#pragma unroll
  for (int s = 0; s < 4; ++s) {
#pragma unroll
    for (int r = 0; r < 8; ++r) {
      int m = r + half * 8;                       // batch row 0..15
      lds[m * 512 + lbase + s * 16 + nin] = acc[s][r];
    }
  }
  __syncthreads();

  // pointwise: 16 batches x 128 hidden units = 2048 items / 256 threads
#pragma unroll
  for (int it = 0; it < 8; ++it) {
    int idx = it * 256 + threadIdx.x;
    int b   = idx >> 7;
    int jj  = idx & 127;
    int jg  = jb * 128 + jj;
    size_t xg = ((size_t)(b * S_ + t)) * G4H_;
    float gi = lds[b * 512 +   0 + jj] + Xg[xg +        jg] + b_ih[       jg] + b_hh[       jg];
    float gf = lds[b * 512 + 128 + jj] + Xg[xg + 1024 + jg] + b_ih[1024 + jg] + b_hh[1024 + jg];
    float gg = lds[b * 512 + 256 + jj] + Xg[xg + 2048 + jg] + b_ih[2048 + jg] + b_hh[2048 + jg];
    float go = lds[b * 512 + 384 + jj] + Xg[xg + 3072 + jg] + b_ih[3072 + jg] + b_hh[3072 + jg];
    float si = 1.f / (1.f + __expf(-gi));
    float sf = 1.f / (1.f + __expf(-gf));
    float so = 1.f / (1.f + __expf(-go));
    int ci = b * H_ + jg;
    float cn = sf * c_st[ci] + si * tanhf(gg);
    c_st[ci] = cn;
    float h = so * tanhf(cn);
    enc[((size_t)(b * S_ + t)) * H_ + jg] = h;
    h_out[ci] = f32_to_bf16_rne(h);
  }
}

// ---------------- span mean pooling + layernorm ---------------------------------
__global__ __launch_bounds__(256)
void k_pool_ln(const float* __restrict__ enc, const int* __restrict__ head,
               const int* __restrict__ tail, const float* __restrict__ gamma,
               const float* __restrict__ beta, float* __restrict__ mv,
               unsigned short* __restrict__ mv_bf) {
  __shared__ float ra[256], rb[256];
  int row = blockIdx.x;                 // b*NS + n
  int b = row / NS_;
  int hd = head[row], tl = tail[row];
  float inv = 1.f / (float)(tl - hd - 1);
  float p[4] = {0.f, 0.f, 0.f, 0.f};
  for (int tok = hd + 1; tok < tl; ++tok) {
    const float* e = enc + ((size_t)(b * S_ + tok)) * H_;
#pragma unroll
    for (int i = 0; i < 4; ++i) p[i] += e[threadIdx.x + 256 * i];
  }
  float s = 0.f, ss = 0.f;
#pragma unroll
  for (int i = 0; i < 4; ++i) { p[i] *= inv; s += p[i]; ss += p[i] * p[i]; }
  int tid = threadIdx.x;
  ra[tid] = s; rb[tid] = ss; __syncthreads();
  for (int st = 128; st > 0; st >>= 1) {
    if (tid < st) { ra[tid] += ra[tid + st]; rb[tid] += rb[tid + st]; }
    __syncthreads();
  }
  float mu = ra[0] * (1.f / H_);
  float var = rb[0] * (1.f / H_) - mu * mu;
  float rs = rsqrtf(var + LN_EPS_);
#pragma unroll
  for (int i = 0; i < 4; ++i) {
    int h = threadIdx.x + 256 * i;
    float y = (p[i] - mu) * rs * gamma[h] + beta[h];
    mv[(size_t)row * H_ + h] = y;
    mv_bf[(size_t)row * H_ + h] = f32_to_bf16_rne(y);
  }
}

// ---------------- attention core (NS=32 spans, per (b,head)) --------------------
__global__ __launch_bounds__(32)
void k_attn(const float* __restrict__ q, const float* __restrict__ k,
            const float* __restrict__ v, const int* __restrict__ mask,
            float* __restrict__ ctx) {
  __shared__ float qs[32][65], ks[32][65], vs[32][65];
  __shared__ int ms[32];
  int b = blockIdx.x / HEADS_, hh = blockIdx.x % HEADS_;
  int qi = threadIdx.x;                // 0..31
  size_t base = ((size_t)(b * NS_ + qi)) * H_ + hh * DH_;
  for (int d = 0; d < DH_; ++d) {
    qs[qi][d] = q[base + d];
    ks[qi][d] = k[base + d];
    vs[qi][d] = v[base + d];
  }
  ms[qi] = mask[b * NS_ + qi];
  __syncthreads();

  int mq = ms[qi];
  float sc[32];
  float mx = -3.0e38f;
#pragma unroll
  for (int kj = 0; kj < 32; ++kj) {
    float s = 0.f;
    for (int d = 0; d < DH_; ++d) s += qs[qi][d] * ks[kj][d];
    s *= 0.125f;                               // 1/sqrt(64)
    bool act = (mq != 0) && (ms[kj] != 0);
    sc[kj] = act ? s : -3.0e38f;
    if (act && s > mx) mx = s;
  }
  float den = 0.f;
#pragma unroll
  for (int kj = 0; kj < 32; ++kj) {
    float e = (sc[kj] > -1.0e37f) ? __expf(sc[kj] - mx) : 0.f;
    sc[kj] = e; den += e;
  }
  float inv = den > 0.f ? 1.f / den : 0.f;
  for (int d = 0; d < DH_; ++d) {
    float a = 0.f;
#pragma unroll
    for (int kj = 0; kj < 32; ++kj) a += sc[kj] * vs[kj][d];
    ctx[base + d] = a * inv;
  }
}

// ---------------- residual + layernorm ------------------------------------------
__global__ __launch_bounds__(256)
void k_res_ln(const float* __restrict__ proj, const float* __restrict__ resid,
              const float* __restrict__ gamma, const float* __restrict__ beta,
              float* __restrict__ out) {
  __shared__ float ra[256], rb[256];
  int row = blockIdx.x;
  float p[4];
  float s = 0.f, ss = 0.f;
#pragma unroll
  for (int i = 0; i < 4; ++i) {
    int h = threadIdx.x + 256 * i;
    p[i] = proj[(size_t)row * H_ + h] + resid[(size_t)row * H_ + h];
    s += p[i]; ss += p[i] * p[i];
  }
  int tid = threadIdx.x;
  ra[tid] = s; rb[tid] = ss; __syncthreads();
  for (int st = 128; st > 0; st >>= 1) {
    if (tid < st) { ra[tid] += ra[tid + st]; rb[tid] += rb[tid + st]; }
    __syncthreads();
  }
  float mu = ra[0] * (1.f / H_);
  float var = rb[0] * (1.f / H_) - mu * mu;
  float rs = rsqrtf(var + LN_EPS_);
#pragma unroll
  for (int i = 0; i < 4; ++i) {
    int h = threadIdx.x + 256 * i;
    out[(size_t)row * H_ + h] = (p[i] - mu) * rs * gamma[h] + beta[h];
  }
}

// ---------------- classifier (L=5) ----------------------------------------------
__global__ __launch_bounds__(256)
void k_clf(const float* __restrict__ x, const float* __restrict__ Wc,
           const float* __restrict__ bc, float* __restrict__ out) {
  __shared__ float red[256];
  int row = blockIdx.x;
  const float* xr = x + (size_t)row * H_;
  for (int l = 0; l < L_; ++l) {
    float s = 0.f;
    for (int h = threadIdx.x; h < H_; h += 256) s += xr[h] * Wc[l * H_ + h];
    red[threadIdx.x] = s; __syncthreads();
    for (int st = 128; st > 0; st >>= 1) {
      if (threadIdx.x < st) red[threadIdx.x] += red[threadIdx.x + st];
      __syncthreads();
    }
    if (threadIdx.x == 0) out[row * L_ + l] = red[0] + bc[l];
    __syncthreads();
  }
}

// ================================================================================
extern "C" void kernel_launch(void* const* d_in, const int* in_sizes, int n_in,
                              void* d_out, int out_size, void* d_ws, size_t ws_size,
                              hipStream_t stream) {
  (void)in_sizes; (void)n_in; (void)out_size; (void)ws_size;
  const float* x    = (const float*)d_in[0];
  const float* Wih  = (const float*)d_in[1];
  const float* Whh  = (const float*)d_in[2];
  const float* bih  = (const float*)d_in[3];
  const float* bhh  = (const float*)d_in[4];
  const float* lng  = (const float*)d_in[5];
  const float* lnb  = (const float*)d_in[6];
  const float* Wq   = (const float*)d_in[7];
  const float* bq   = (const float*)d_in[8];
  const float* Wk   = (const float*)d_in[9];
  const float* bk   = (const float*)d_in[10];
  const float* Wv   = (const float*)d_in[11];
  const float* bv   = (const float*)d_in[12];
  const float* Wo   = (const float*)d_in[13];
  const float* bo   = (const float*)d_in[14];
  const float* alng = (const float*)d_in[15];
  const float* alnb = (const float*)d_in[16];
  const float* Wc   = (const float*)d_in[17];
  const float* bc   = (const float*)d_in[18];
  const int* sphead = (const int*)d_in[19];
  const int* sptail = (const int*)d_in[20];
  const int* spmask = (const int*)d_in[21];
  float* logits = (float*)d_out;

  // workspace layout
  char* ws = (char*)d_ws;
  size_t off = 0;
  auto carve = [&](size_t bytes) -> char* {
    char* p = ws + off;
    off = (off + bytes + 255) & ~(size_t)255;
    return p;
  };
  const size_t BS = (size_t)B_ * S_;          // 16384
  const size_t BN = (size_t)B_ * NS_;         // 512
  float*          Xg     = (float*)         carve(BS * G4H_ * sizeof(float));       // 256 MB
  float*          enc    = (float*)         carve(BS * H_ * sizeof(float));         // 64 MB
  unsigned short* x_bf   = (unsigned short*)carve(BS * H_ * sizeof(unsigned short));
  unsigned short* wih_bf = (unsigned short*)carve((size_t)G4H_ * H_ * 2);
  unsigned short* whh_bf = (unsigned short*)carve((size_t)G4H_ * H_ * 2);
  unsigned short* wq_bf  = (unsigned short*)carve((size_t)H_ * H_ * 2);
  unsigned short* wk_bf  = (unsigned short*)carve((size_t)H_ * H_ * 2);
  unsigned short* wv_bf  = (unsigned short*)carve((size_t)H_ * H_ * 2);
  unsigned short* wo_bf  = (unsigned short*)carve((size_t)H_ * H_ * 2);
  unsigned short* h_bf   = (unsigned short*)carve((size_t)2 * B_ * H_ * 2);         // ping-pong
  float*          c_st   = (float*)         carve((size_t)B_ * H_ * sizeof(float));
  float*          mv     = (float*)         carve(BN * H_ * sizeof(float));
  unsigned short* mv_bf  = (unsigned short*)carve(BN * H_ * 2);
  float*          qf     = (float*)         carve(BN * H_ * sizeof(float));
  float*          kf     = (float*)         carve(BN * H_ * sizeof(float));
  float*          vf     = (float*)         carve(BN * H_ * sizeof(float));
  float*          ctxf   = (float*)         carve(BN * H_ * sizeof(float));
  unsigned short* ctx_bf = (unsigned short*)carve(BN * H_ * 2);
  float*          projf  = (float*)         carve(BN * H_ * sizeof(float));
  float*          aof    = (float*)         carve(BN * H_ * sizeof(float));

  auto cvt = [&](const float* src, unsigned short* dst, size_t n) {
    k_f32_to_bf16<<<(unsigned)((n + 255) / 256), 256, 0, stream>>>(src, dst, (int)n);
  };

  // 1) bf16 conversions of activations + weights
  cvt(x,   x_bf,   BS * H_);
  cvt(Wih, wih_bf, (size_t)G4H_ * H_);
  cvt(Whh, whh_bf, (size_t)G4H_ * H_);
  cvt(Wq,  wq_bf,  (size_t)H_ * H_);
  cvt(Wk,  wk_bf,  (size_t)H_ * H_);
  cvt(Wv,  wv_bf,  (size_t)H_ * H_);
  cvt(Wo,  wo_bf,  (size_t)H_ * H_);

  // 2) state init (c = 0, h ping = 0)
  k_init_state<<<(B_ * H_ + 255) / 256, 256, 0, stream>>>(c_st, h_bf, B_ * H_);

  // 3) input projection for all timesteps: Xg = X @ Wih^T  (bias folded in later)
  {
    dim3 grid(G4H_ / 256, (unsigned)(BS / 16));
    k_gemm_bf16_nt<<<grid, 128, 0, stream>>>(x_bf, wih_bf, Xg, nullptr,
                                             (int)BS, G4H_, H_);
  }

  // 4) sequential LSTM: one fused WMMA+pointwise kernel per timestep
  for (int t = 0; t < S_; ++t) {
    const unsigned short* hin = h_bf + (size_t)(t & 1) * B_ * H_;
    unsigned short* hout      = h_bf + (size_t)((t + 1) & 1) * B_ * H_;
    k_lstm_step<<<8, 256, 0, stream>>>(hin, whh_bf, Xg, bih, bhh,
                                       c_st, enc, hout, t);
  }

  // 5) span mean pooling + layernorm
  k_pool_ln<<<(unsigned)BN, 256, 0, stream>>>(enc, sphead, sptail, lng, lnb, mv, mv_bf);

  // 6) Q/K/V projections (WMMA)
  {
    dim3 grid(H_ / 256, (unsigned)(BN / 16));
    k_gemm_bf16_nt<<<grid, 128, 0, stream>>>(mv_bf, wq_bf, qf, bq, (int)BN, H_, H_);
    k_gemm_bf16_nt<<<grid, 128, 0, stream>>>(mv_bf, wk_bf, kf, bk, (int)BN, H_, H_);
    k_gemm_bf16_nt<<<grid, 128, 0, stream>>>(mv_bf, wv_bf, vf, bv, (int)BN, H_, H_);
  }

  // 7) attention core per (batch, head)
  k_attn<<<B_ * HEADS_, 32, 0, stream>>>(qf, kf, vf, spmask, ctxf);

  // 8) output projection (WMMA) + residual + layernorm
  cvt(ctxf, ctx_bf, BN * H_);
  {
    dim3 grid(H_ / 256, (unsigned)(BN / 16));
    k_gemm_bf16_nt<<<grid, 128, 0, stream>>>(ctx_bf, wo_bf, projf, bo, (int)BN, H_, H_);
  }
  k_res_ln<<<(unsigned)BN, 256, 0, stream>>>(projf, mv, alng, alnb, aof);

  // 9) classifier -> logits
  k_clf<<<(unsigned)BN, 256, 0, stream>>>(aof, Wc, bc, logits);
}
